// ContrastiveEGNN_78065325572484
// MI455X (gfx1250) — matrix operand order repaired and checked by reading, not compiled, not run on hardware
//
#include <hip/hip_runtime.h>
#include <hip/hip_fp16.h>

typedef _Float16 f16;
typedef __attribute__((ext_vector_type(16))) _Float16 v16h;
typedef __attribute__((ext_vector_type(8)))  _Float16 f16x8;
typedef __attribute__((ext_vector_type(2)))  _Float16 f16x2;
typedef __attribute__((ext_vector_type(8)))  float    v8f;
typedef __attribute__((ext_vector_type(4)))  unsigned int u32x4;
typedef __attribute__((ext_vector_type(8)))  int      i32x8;
typedef __attribute__((ext_vector_type(4)))  int      i32x4;

#define NNODES   10000
#define NEDGES   160000
#define NGRAPHS  32
#define IN_NF    960
#define HID      512
#define HID2     128
#define EDGE_NFD 11
#define PROJ     96
#define OUTD     64

#define MT   64
#define NT   64
#define LSTR 40   // LDS row stride in halfs: 80B rows -> 16B-aligned stores, conflict-free frags

#if defined(__has_builtin)
#if __has_builtin(__builtin_amdgcn_tensor_load_to_lds) && __has_builtin(__builtin_amdgcn_s_wait_tensorcnt)
#define HAVE_TDM 1
#endif
#endif
#ifndef HAVE_TDM
#define HAVE_TDM 0
#endif

#if HAVE_TDM
// TDM load of a 64-row x 32-half f16 tile into LDS, padded 16B per 64B row
// so the LDS image has row stride LSTR(=40) halfs.  D# layout per ISA ch.8.
__device__ __forceinline__ void tdm_load_tile(const f16* gsrc, unsigned lds_off,
                                              int K, int rows_left) {
    const unsigned long long ga = (unsigned long long)(size_t)gsrc;
    u32x4 g0;
    g0[0] = 1u;                                              // count=1, user D#
    g0[1] = lds_off;                                         // lds_addr (bytes)
    g0[2] = (unsigned)(ga & 0xFFFFFFFFu);                    // global_addr lo
    g0[3] = (unsigned)((ga >> 32) & 0x01FFFFFFu) | (2u << 30); // addr hi | type=2
    i32x8 g1;
    g1[0] = (1 << 16) | (1 << 20) | (3 << 22) | (3 << 25);   // 2B elems, pad_en, 64B interval, +4 DW
    g1[1] = (K & 0xFFFF) << 16;                              // tensor_dim0[15:0]
    g1[2] = ((K >> 16) & 0xFFFF) | ((rows_left & 0xFFFF) << 16); // dim0 hi | dim1 lo
    g1[3] = ((rows_left >> 16) & 0xFFFF) | (32 << 16);       // dim1 hi | tile_dim0=32
    g1[4] = 64;                                              // tile_dim1=64, tile_dim2=0
    g1[5] = K;                                               // tensor_dim0_stride (elems)
    g1[6] = 0;
    g1[7] = 0;
    i32x4 z4 = {0, 0, 0, 0};
#if __clang_major__ >= 23
    i32x8 z8 = {0, 0, 0, 0, 0, 0, 0, 0};
    __builtin_amdgcn_tensor_load_to_lds(g0, g1, z4, z4, z8, 0);
#else
    __builtin_amdgcn_tensor_load_to_lds(g0, g1, z4, z4, 0);
#endif
}
#endif

// =====================================================================
// One (A,W) accumulation pass over K, fully specialized at compile time.
// Tile: 64x64 per 256-thread block; 8 waves; wave = 16 rows x 32 cols
// (one A fragment reused by two v_wmma_f32_16x16x32_f16).
// TDMA: stage the A tile with the Tensor Data Mover instead of VALU.
// =====================================================================
template<bool AF16, bool GATHER, bool TDMA>
__device__ __forceinline__ void gemm_pass(
    const void* A, const int* idx, const f16* W,
    int K, int M, int N, int tileM, int n0,
    f16* Asub, f16* Bt, v8f& acc0, v8f& acc1)
{
    const int tid = threadIdx.x;
    const int ar  = tid >> 2;          // A stage: row 0..63
    const int ak  = (tid & 3) << 3;    //          k 0,8,16,24
    const int bk  = tid >> 3;          // B stage: k 0..31
    const int bn  = (tid & 7) << 3;    //          n 0,8,..,56

    const int wave = tid >> 5;
    const int lane = tid & 31;
    const int wm   = (wave >> 1) << 4; // 0,16,32,48
    const int wn   = (wave & 1) << 5;  // 0,32
    const int hsel = lane >> 4;
    const int l16  = lane & 15;

    // branch-free A row (OOB rows clamped; their outputs are never stored)
    const int gmc  = (tileM + ar < M) ? (tileM + ar) : (M - 1);
    const int arow = GATHER ? idx[gmc] : gmc;

    for (int kk = 0; kk < K; kk += 32) {
        // ---- stage A tile ----
        if (TDMA && HAVE_TDM) {
#if HAVE_TDM
            if (wave == 0)
                tdm_load_tile((const f16*)A + (size_t)tileM * K + kk,
                              (unsigned)(size_t)(void*)Asub, K, M - tileM);
#endif
        } else {
            f16x8 av;
            if (AF16) {
                av = *(const f16x8*)((const f16*)A + (size_t)arow * K + kk + ak);
            } else {
                const float* Ap = (const float*)A + (size_t)arow * K + kk + ak;
                #pragma unroll
                for (int j = 0; j < 8; ++j) av[j] = (f16)Ap[j];
            }
            *(f16x8*)&Asub[ar * LSTR + ak] = av;
        }
        // ---- stage B tile transposed: Bt[n][k] = W[(kk+k)*N + n0+n] ----
        {
            f16x8 wv = *(const f16x8*)(W + (size_t)(kk + bk) * N + n0 + bn);
            if (n0 + NT > N) {           // block-uniform tail tile only
                #pragma unroll
                for (int j = 0; j < 8; ++j)
                    if (n0 + bn + j >= N) wv[j] = (f16)0.f;
            }
            #pragma unroll
            for (int j = 0; j < 8; ++j) Bt[(bn + j) * LSTR + bk] = wv[j];
        }
#if HAVE_TDM
        if (TDMA) { if (wave == 0) __builtin_amdgcn_s_wait_tensorcnt(0); }
#endif
        __syncthreads();

        // prefetch next weight k-slab (gfx1250 global_prefetch_b8 path)
        if (kk + 32 < K)
            __builtin_prefetch(W + (size_t)(kk + 32 + bk) * N + n0 + bn, 0, 1);

        // ---- fragments (ISA 7.12.2 16-bit layouts) ----
        v16h afrag, bfrag0, bfrag1;
        f16x2* a2 = (f16x2*)&afrag;
        f16x2* b0 = (f16x2*)&bfrag0;
        f16x2* b1 = (f16x2*)&bfrag1;
        #pragma unroll
        for (int v = 0; v < 8; ++v) {
            // A 16x32: v<4 -> K = hsel*8 + 2v ; v>=4 -> 16 + hsel*8 + 2(v-4)
            const int k0 = ((v & 4) << 2) + (hsel << 3) + ((v & 3) << 1);
            a2[v] = *(const f16x2*)&Asub[(wm + l16) * LSTR + k0];
        }
        #pragma unroll
        for (int j = 0; j < 8; ++j) {
            // B 32x16: j<4 -> K = hsel*8 + 2j ; j>=4 -> 16 + hsel*8 + 2(j-4)
            const int kb = ((j & 4) << 2) + (hsel << 3) + ((j & 3) << 1);
            b0[j] = *(const f16x2*)&Bt[(wn + l16) * LSTR + kb];
            b1[j] = *(const f16x2*)&Bt[(wn + 16 + l16) * LSTR + kb];
        }
        acc0 = __builtin_amdgcn_wmma_f32_16x16x32_f16(false, afrag, false, bfrag0,
                                                      (short)0, acc0, false, false);
        acc1 = __builtin_amdgcn_wmma_f32_16x16x32_f16(false, afrag, false, bfrag1,
                                                      (short)0, acc1, false, false);
        __syncthreads();
    }
}

// out[M,N] = act( sum_src A_src[idx][K] @ W_src[K,N] + smalltail + bias ) + residual
template<bool TWO, bool A0F16, bool A1F16, bool G0, bool G1, bool SMALL>
__global__ __launch_bounds__(256) void wmma_gemm_kernel(
    const void* A0, const int* idx0, const f16* W0,
    const void* A1, const int* idx1, const f16* W1,
    int K,
    const float* Wsm, const float* radial, const float* ea,  // K=12 f32 tail (e1)
    const float* bias, int act,                              // 0 none, 1 silu, 2 relu
    const void* res, int resf16,
    void* out, int outf16,
    int M, int N)
{
    __shared__ __align__(16) f16 Asub[MT * LSTR];
    __shared__ __align__(16) f16 Bt[NT * LSTR];

    const int tid   = threadIdx.x;
    const int tileM = blockIdx.x * MT;
    const int n0    = blockIdx.y * NT;

    const int wave = tid >> 5;
    const int lane = tid & 31;
    const int wm   = (wave >> 1) << 4;
    const int wn   = (wave & 1) << 5;
    const int hsel = lane >> 4;
    const int l16  = lane & 15;

    v8f acc0, acc1;
    #pragma unroll
    for (int i = 0; i < 8; ++i) { acc0[i] = 0.f; acc1[i] = 0.f; }

    gemm_pass<A0F16, G0, (A0F16 && !G0)>(A0, idx0, W0, K, M, N, tileM, n0, Asub, Bt, acc0, acc1);
    if (TWO)
        gemm_pass<A1F16, G1, (A1F16 && !G1)>(A1, idx1, W1, K, M, N, tileM, n0, Asub, Bt, acc0, acc1);

    // C/D layout: reg r -> m = hsel*8 + r, n = l16
    const int cmb = wm + (hsel << 3);
    #pragma unroll
    for (int cset = 0; cset < 2; ++cset) {
        v8f& acc = cset ? acc1 : acc0;
        const int cn = n0 + wn + (cset << 4) + l16;
        if (cn >= N) continue;
        if (SMALL) {   // f32 tail: row 0 = radial, rows 1..11 = edge_attr
            #pragma unroll
            for (int r = 0; r < 8; ++r) {
                const int gm = tileM + cmb + r;
                if (gm < M) {
                    float v = radial[gm] * Wsm[cn];
                    const float* er = ea + (size_t)gm * EDGE_NFD;
                    #pragma unroll
                    for (int j = 0; j < EDGE_NFD; ++j)
                        v += er[j] * Wsm[(size_t)(1 + j) * N + cn];
                    acc[r] += v;
                }
            }
        }
        const float bv = bias ? bias[cn] : 0.f;
        #pragma unroll
        for (int r = 0; r < 8; ++r) {
            const int gm = tileM + cmb + r;
            if (gm >= M) continue;
            float v = acc[r] + bv;
            if (act == 1)      v = v / (1.f + __expf(-v));   // SiLU
            else if (act == 2) v = fmaxf(v, 0.f);            // ReLU
            if (res) {
                v += resf16 ? (float)((const f16*)res)[(size_t)gm * N + cn]
                            : ((const float*)res)[(size_t)gm * N + cn];
            }
            if (outf16) ((f16*)out)[(size_t)gm * N + cn] = (f16)v;
            else        ((float*)out)[(size_t)gm * N + cn] = v;
        }
    }
}

// =====================================================================
// Elementwise / reduction helpers
// =====================================================================
__global__ void cvt_f32_f16_kernel(const float* s, f16* d, int n) {
    const int i = blockIdx.x * 256 + threadIdx.x;
    if (i < n) d[i] = (f16)s[i];
}

__global__ void edge_geom_kernel(const float* x, const int* row, const int* col,
                                 float* diff, float* radial, int E) {
    const int e = blockIdx.x * 256 + threadIdx.x;
    if (e >= E) return;
    const int r = row[e], c = col[e];
    const float dx = x[r*3+0] - x[c*3+0];
    const float dy = x[r*3+1] - x[c*3+1];
    const float dz = x[r*3+2] - x[c*3+2];
    const float rad = dx*dx + dy*dy + dz*dz;
    const float inv = 1.f / (sqrtf(rad) + 1e-8f);
    diff[e*3+0] = dx*inv; diff[e*3+1] = dy*inv; diff[e*3+2] = dz*inv;
    radial[e] = rad;
}

// m *= sigmoid(m @ attW + attB)   (wave per edge, in place)
__global__ __launch_bounds__(256) void att_gate_kernel(f16* m, const float* attW,
                                                       const float* attB, int E, int H) {
    const int e    = (blockIdx.x * 256 + threadIdx.x) >> 5;
    const int lane = threadIdx.x & 31;
    if (e >= E) return;
    f16* mr = m + (size_t)e * H;
    float dot = 0.f;
    for (int k = lane; k < H; k += 32) dot += (float)mr[k] * attW[k];
    #pragma unroll
    for (int off = 16; off > 0; off >>= 1) dot += __shfl_xor(dot, off, 32);
    const float sg = 1.f / (1.f + __expf(-(dot + attB[0])));
    for (int k = lane; k < H; k += 32) mr[k] = (f16)((float)mr[k] * sg);
}

// s = t[e] . c2W ; atomic coordsum[row[e]] += diff[e] * s   (wave per edge)
__global__ __launch_bounds__(256) void coord_kernel(const f16* t, const float* c2W,
                                                    const float* diff, const int* row,
                                                    float* csum, int E, int H) {
    const int e    = (blockIdx.x * 256 + threadIdx.x) >> 5;
    const int lane = threadIdx.x & 31;
    if (e >= E) return;
    const f16* tr = t + (size_t)e * H;
    float dot = 0.f;
    for (int k = lane; k < H; k += 32) dot += (float)tr[k] * c2W[k];
    #pragma unroll
    for (int off = 16; off > 0; off >>= 1) dot += __shfl_xor(dot, off, 32);
    if (lane < 3)
        atomicAdd(&csum[(size_t)row[e] * 3 + lane], diff[(size_t)e * 3 + lane] * dot);
}

__global__ void count_kernel(const int* idx, float* cnt, int n) {
    const int i = blockIdx.x * 256 + threadIdx.x;
    if (i < n) atomicAdd(&cnt[idx[i]], 1.f);
}

__global__ void coord_update_kernel(float* x, const float* csum, const float* cnt, int nn) {
    const int t = blockIdx.x * 256 + threadIdx.x;
    if (t >= nn * 3) return;
    x[t] += csum[t] / fmaxf(cnt[t / 3], 1.f);
}

__global__ void scatter_add_kernel(const f16* m, const int* row, float* agg, int E, int H) {
    const long t   = (long)blockIdx.x * 256 + threadIdx.x;
    const int  per = H >> 2;
    if (t >= (long)E * per) return;
    const int e = (int)(t / per);
    const int k = (int)(t % per) << 2;
    const int n = row[e];
    const f16* mp = m + (size_t)e * H + k;
    float*     ap = agg + (size_t)n * H + k;
    #pragma unroll
    for (int j = 0; j < 4; ++j) atomicAdd(&ap[j], (float)mp[j]);
}

__global__ void pool_kernel(const f16* h, const int* batch, float* gsum, int nn, int H) {
    const int t = blockIdx.x * 256 + threadIdx.x;
    if (t >= nn * H) return;
    atomicAdd(&gsum[(size_t)batch[t / H] * H + (t % H)], (float)h[t]);
}

__global__ void pool_div_kernel(const float* gsum, const float* cnt, float* g, int H, int ng) {
    const int t = blockIdx.x * 256 + threadIdx.x;
    if (t >= ng * H) return;
    g[t] = gsum[t] / fmaxf(cnt[t / H], 1.f);
}

// =====================================================================
// Host orchestration
// =====================================================================
struct WsAlloc {
    char* base; size_t cur;
    void* take(size_t bytes) {
        void* p = base + cur;
        cur = (cur + bytes + 255) & ~(size_t)255;
        return p;
    }
};

static const f16* to_f16(hipStream_t s, WsAlloc& ws, const float* src, size_t n) {
    f16* d = (f16*)ws.take(n * sizeof(f16));
    cvt_f32_f16_kernel<<<(int)((n + 255) / 256), 256, 0, s>>>(src, d, (int)n);
    return d;
}

static void launch_gemm(hipStream_t s,
    const void* A0, int a0f16, const int* idx0, const f16* W0,
    const void* A1, int a1f16, const int* idx1, const f16* W1,
    int K, const float* Wsm, const float* radial, const float* ea,
    const float* bias, int act, const void* res, int resf16,
    void* out, int outf16, int M, int N)
{
    (void)a1f16;
    dim3 g((M + MT - 1) / MT, (N + NT - 1) / NT);
    if (A1 == nullptr) {
        if (a0f16)
            wmma_gemm_kernel<false,true,false,false,false,false><<<g,256,0,s>>>(
                A0, idx0, W0, A1, idx1, W1, K, Wsm, radial, ea, bias, act, res, resf16, out, outf16, M, N);
        else
            wmma_gemm_kernel<false,false,false,false,false,false><<<g,256,0,s>>>(
                A0, idx0, W0, A1, idx1, W1, K, Wsm, radial, ea, bias, act, res, resf16, out, outf16, M, N);
    } else if (Wsm) {   // e1: dual gathered f16 sources + f32 small tail
        wmma_gemm_kernel<true,true,true,true,true,true><<<g,256,0,s>>>(
            A0, idx0, W0, A1, idx1, W1, K, Wsm, radial, ea, bias, act, res, resf16, out, outf16, M, N);
    } else {            // n1: f16 + f32 sources, no gather
        wmma_gemm_kernel<true,true,false,false,false,false><<<g,256,0,s>>>(
            A0, idx0, W0, A1, idx1, W1, K, Wsm, radial, ea, bias, act, res, resf16, out, outf16, M, N);
    }
}

struct Ctx {
    hipStream_t s;
    const int *row, *col;
    const float* ea;
    float *cnt_row, *diffb, *radialb, *coordsum, *agg;
    f16 *EA, *EB, *nhT;
};

static void run_gcl(const Ctx& c, f16* h, float* x, int H,
                    const f16* e1W16, const float* e1Wf, const float* e1B,
                    const f16* e2W16, const float* e2B,
                    const float* attW, const float* attB,
                    const f16* c1W16, const float* c1B, const float* c2W,
                    const f16* n1W16, const float* n1B,
                    const f16* n2W16, const float* n2B)
{
    const int E = NEDGES, NN = NNODES;
    hipStream_t s = c.s;

    edge_geom_kernel<<<(E + 255) / 256, 256, 0, s>>>(x, c.row, c.col, c.diffb, c.radialb, E);

    // m1 = silu(e1([h[row], h[col], radial, ea]))  -> EA
    launch_gemm(s, h, 1, c.row, e1W16,
                   h, 1, c.col, e1W16 + (size_t)H * H, H,
                e1Wf + (size_t)2 * H * H, c.radialb, c.ea,
                e1B, 1, nullptr, 0, c.EA, 1, E, H);
    // m2 = silu(e2(m1)) -> EB
    launch_gemm(s, c.EA, 1, nullptr, e2W16, nullptr, 0, nullptr, nullptr, H,
                nullptr, nullptr, nullptr, e2B, 1, nullptr, 0, c.EB, 1, E, H);
    // gate in place on EB
    att_gate_kernel<<<(E * 32 + 255) / 256, 256, 0, s>>>(c.EB, attW, attB, E, H);
    // c1 = silu(c1(m)) -> EA
    launch_gemm(s, c.EB, 1, nullptr, c1W16, nullptr, 0, nullptr, nullptr, H,
                nullptr, nullptr, nullptr, c1B, 1, nullptr, 0, c.EA, 1, E, H);
    // coord update: x += seg_mean(diff * (c1out @ c2W), row)
    hipMemsetAsync(c.coordsum, 0, (size_t)NN * 3 * sizeof(float), s);
    coord_kernel<<<(E * 32 + 255) / 256, 256, 0, s>>>(c.EA, c2W, c.diffb, c.row, c.coordsum, E, H);
    coord_update_kernel<<<(NN * 3 + 255) / 256, 256, 0, s>>>(x, c.coordsum, c.cnt_row, NN);
    // agg = segment_sum(m, row)
    hipMemsetAsync(c.agg, 0, (size_t)NN * H * sizeof(float), s);
    {
        long total = (long)E * (H >> 2);
        scatter_add_kernel<<<(int)((total + 255) / 256), 256, 0, s>>>(c.EB, c.row, c.agg, E, H);
    }
    // t = silu(n1([h, agg])) -> nhT
    launch_gemm(s, h, 1, nullptr, n1W16,
                   c.agg, 0, nullptr, n1W16 + (size_t)H * H, H,
                nullptr, nullptr, nullptr, n1B, 1, nullptr, 0, c.nhT, 1, NN, H);
    // h = h + n2(t)   (residual fused, in place)
    launch_gemm(s, c.nhT, 1, nullptr, n2W16, nullptr, 0, nullptr, nullptr, H,
                nullptr, nullptr, nullptr, n2B, 0, h, 1, h, 1, NN, H);
}

// Param leaf layout (insertion-order flatten of setup_inputs()):
//   egnn base b: b+0 emb_in.W, b+1 emb_in.b,
//   layer L at lb=b+2+13L: e1.W,e1.b,e2.W,e2.b,att.W,att.b,c1.W,c1.b,c2W,n1.W,n1.b,n2.W,n2.b
//   b+54 emb_out.W, b+55 emb_out.b
static void run_egnn(const Ctx& c, WsAlloc& ws, void* const* d_in, int base,
                     int din, int H, int dout,
                     const void* h_in, int h_in_f16, float* x, f16* h_work, f16* h_out)
{
    hipStream_t s = c.s;
    const float* embinW  = (const float*)d_in[base + 0];
    const float* embinB  = (const float*)d_in[base + 1];
    const float* emboutW = (const float*)d_in[base + 54];
    const float* emboutB = (const float*)d_in[base + 55];

    const f16* embinW16  = to_f16(s, ws, embinW,  (size_t)din * H);
    const f16* emboutW16 = to_f16(s, ws, emboutW, (size_t)H * dout);

    // h = emb_in(h_in)
    launch_gemm(s, h_in, h_in_f16, nullptr, embinW16, nullptr, 0, nullptr, nullptr, din,
                nullptr, nullptr, nullptr, embinB, 0, nullptr, 0, h_work, 1, NNODES, H);

    for (int L = 0; L < 4; ++L) {
        const int lb = base + 2 + 13 * L;
        const float* e1W  = (const float*)d_in[lb + 0];
        const float* e1B  = (const float*)d_in[lb + 1];
        const float* e2W  = (const float*)d_in[lb + 2];
        const float* e2B  = (const float*)d_in[lb + 3];
        const float* attW = (const float*)d_in[lb + 4];
        const float* attB = (const float*)d_in[lb + 5];
        const float* c1W  = (const float*)d_in[lb + 6];
        const float* c1B  = (const float*)d_in[lb + 7];
        const float* c2W  = (const float*)d_in[lb + 8];
        const float* n1W  = (const float*)d_in[lb + 9];
        const float* n1B  = (const float*)d_in[lb + 10];
        const float* n2W  = (const float*)d_in[lb + 11];
        const float* n2B  = (const float*)d_in[lb + 12];

        const f16* e1W16 = to_f16(s, ws, e1W, (size_t)(2 * H + 1 + EDGE_NFD) * H);
        const f16* e2W16 = to_f16(s, ws, e2W, (size_t)H * H);
        const f16* c1W16 = to_f16(s, ws, c1W, (size_t)H * H);
        const f16* n1W16 = to_f16(s, ws, n1W, (size_t)2 * H * H);
        const f16* n2W16 = to_f16(s, ws, n2W, (size_t)H * H);

        run_gcl(c, h_work, x, H, e1W16, e1W, e1B, e2W16, e2B, attW, attB,
                c1W16, c1B, c2W, n1W16, n1B, n2W16, n2B);
    }

    // h_out = emb_out(h)
    launch_gemm(s, h_work, 1, nullptr, emboutW16, nullptr, 0, nullptr, nullptr, H,
                nullptr, nullptr, nullptr, emboutB, 0, nullptr, 0, h_out, 1, NNODES, dout);
}

extern "C" void kernel_launch(void* const* d_in, const int* in_sizes, int n_in,
                              void* d_out, int out_size, void* d_ws, size_t ws_size,
                              hipStream_t stream) {
    (void)in_sizes; (void)n_in; (void)out_size; (void)ws_size;

    const float* node_features = (const float*)d_in[0];
    const int*   edge_index    = (const int*)d_in[1];
    const float* node_pos      = (const float*)d_in[2];
    const float* edge_attr     = (const float*)d_in[3];
    const int*   batch         = (const int*)d_in[4];
    const int* row = edge_index;
    const int* col = edge_index + NEDGES;

    WsAlloc ws{(char*)d_ws, 0};
    f16*   EA       = (f16*)ws.take((size_t)NEDGES * HID * sizeof(f16));
    f16*   EB       = (f16*)ws.take((size_t)NEDGES * HID * sizeof(f16));
    float* diffb    = (float*)ws.take((size_t)NEDGES * 3 * sizeof(float));
    float* radialb  = (float*)ws.take((size_t)NEDGES * sizeof(float));
    float* agg      = (float*)ws.take((size_t)NNODES * HID * sizeof(float));
    float* coordsum = (float*)ws.take((size_t)NNODES * 3 * sizeof(float));
    float* cnt_row  = (float*)ws.take((size_t)NNODES * sizeof(float));
    float* cnt_bat  = (float*)ws.take((size_t)NGRAPHS * sizeof(float));
    float* xbuf     = (float*)ws.take((size_t)NNODES * 3 * sizeof(float));
    f16*   h_work   = (f16*)ws.take((size_t)NNODES * HID * sizeof(f16));
    f16*   nhT      = (f16*)ws.take((size_t)NNODES * HID * sizeof(f16));
    f16*   h_mid    = (f16*)ws.take((size_t)NNODES * HID * sizeof(f16));
    f16*   h_fin    = (f16*)ws.take((size_t)NNODES * HID2 * sizeof(f16));
    float* gsum     = (float*)ws.take((size_t)NGRAPHS * HID2 * sizeof(float));
    float* gbuf     = (float*)ws.take((size_t)NGRAPHS * HID2 * sizeof(float));
    f16*   fe_t     = (f16*)ws.take((size_t)NGRAPHS * PROJ * sizeof(f16));

    // per-node edge counts (constant over all layers) and per-graph node counts
    hipMemsetAsync(cnt_row, 0, (size_t)NNODES * sizeof(float), stream);
    count_kernel<<<(NEDGES + 255) / 256, 256, 0, stream>>>(row, cnt_row, NEDGES);
    hipMemsetAsync(cnt_bat, 0, (size_t)NGRAPHS * sizeof(float), stream);
    count_kernel<<<(NNODES + 255) / 256, 256, 0, stream>>>(batch, cnt_bat, NNODES);

    Ctx c{stream, row, col, edge_attr, cnt_row, diffb, radialb, coordsum, agg, EA, EB, nhT};

    for (int br = 0; br < 2; ++br) {
        const int b1 = 5 + (br ? 112 : 0);    // egnn{1,2}_1
        const int b2 = b1 + 56;               // egnn{1,2}_2
        const int fb = 5 + 224 + br * 4;      // fe{1,2}

        hipMemcpyAsync(xbuf, node_pos, (size_t)NNODES * 3 * sizeof(float),
                       hipMemcpyDeviceToDevice, stream);

        run_egnn(c, ws, d_in, b1, IN_NF, HID,  HID,  node_features, 0, xbuf, h_work, h_mid);
        run_egnn(c, ws, d_in, b2, HID,   HID2, HID2, h_mid,         1, xbuf, h_work, h_fin);

        // g = seg_mean(h_fin, batch)
        hipMemsetAsync(gsum, 0, (size_t)NGRAPHS * HID2 * sizeof(float), stream);
        pool_kernel<<<(NNODES * HID2 + 255) / 256, 256, 0, stream>>>(h_fin, batch, gsum, NNODES, HID2);
        pool_div_kernel<<<(NGRAPHS * HID2 + 255) / 256, 256, 0, stream>>>(gsum, cnt_bat, gbuf, HID2, NGRAPHS);

        // feature extractor: l2(relu(l1(g)))
        const float* l1W = (const float*)d_in[fb + 0];
        const float* l1B = (const float*)d_in[fb + 1];
        const float* l2W = (const float*)d_in[fb + 2];
        const float* l2B = (const float*)d_in[fb + 3];
        const f16* l1W16 = to_f16(stream, ws, l1W, (size_t)HID2 * PROJ);
        const f16* l2W16 = to_f16(stream, ws, l2W, (size_t)PROJ * OUTD);

        launch_gemm(stream, gbuf, 0, nullptr, l1W16, nullptr, 0, nullptr, nullptr, HID2,
                    nullptr, nullptr, nullptr, l1B, 2, nullptr, 0, fe_t, 1, NGRAPHS, PROJ);
        float* outp = (float*)d_out + (size_t)br * NGRAPHS * OUTD;
        launch_gemm(stream, fe_t, 1, nullptr, l2W16, nullptr, 0, nullptr, nullptr, PROJ,
                    nullptr, nullptr, nullptr, l2B, 0, nullptr, 0, outp, 0, NGRAPHS, OUTD);
    }
}
